// Net_22625887715642
// MI455X (gfx1250) — matrix operand order repaired and checked by reading, not compiled, not run on hardware
//
#include <hip/hip_runtime.h>

// ---------------------------------------------------------------------------
// MI455X (gfx1250) implementation.
//   Stage 1: 16x 9x9 conv + channel normalize  (trivial FLOPs)
//   Stage 2: all-pairs template correlation as f16 WMMA GEMM w/ f32 accum.
//            M = 37x37 shift positions (16-wide dy tiles), N = 16 templates,
//            K = 212*212 (rows padded to 224 = 7 chunks of 32).
//   WMMA stage uses LDS staging:
//     - B (templates): double-buffered row (16 x 224 f16), shared by 8 waves
//     - A (maps): 17-row ring buffer, 1 new global row per K-row iteration
// ---------------------------------------------------------------------------

typedef __attribute__((ext_vector_type(8)))  _Float16 v8h;
typedef __attribute__((ext_vector_type(16))) _Float16 v16h;
typedef __attribute__((ext_vector_type(8)))  float    v8f;

#define IN_H   256
#define IN_W   256
#define OH     248
#define OW     248
#define NF     16
#define KH     9
#define KW     9
#define NTAP   (KH*KW)          // 81
#define CORR   18               // num_rfs * 9
#define TSIDE  212              // template side (248 - 2*18)
#define TPAD   224              // K-row padded to 7*32
#define NCHUNK 7                // 224 / 32
#define XP     264              // padded f16 activation side (guards sliding loads)
#define NPOS   37               // 248 - 212 + 1
#define X1_SZ  (NF*OH*OW)       // 984064 floats (first output)
#define XH_ELEMS (NF*XP*XP)
#define BT_ELEMS (NF*TSIDE*TPAD)
#define ASTRIDE 248             // LDS A row stride (f16): 124 dw -> conflict-free
#define ARING   17              // A ring rows (16 live + 1 being filled)
#define BSTRIDE 232             // LDS B row stride (f16): 116 dw -> conflict-free
#define NPAIR   (NF*3)          // 48 (n, dy-tile) pairs
#define BLKPP   5               // blocks per pair (8 dx each, covers 37)

__global__ __launch_bounds__(256) void zero_u32_kernel(unsigned int* __restrict__ p, int n) {
    int i = blockIdx.x * 256 + threadIdx.x;
    if (i < n) p[i] = 0u;
}

// Stage 1: conv (temporal filters averaged over depth, /2 folded in), relu,
// channel-sum normalization. Writes f32 x1f to d_out and f16 copy to d_ws.
__global__ __launch_bounds__(256) void conv_norm_kernel(
    const float* __restrict__ x, const float* __restrict__ ft,
    const float* __restrict__ fn, float* __restrict__ out,
    _Float16* __restrict__ xh)
{
    __shared__ float wlds[NF * NTAP];
    int tid = threadIdx.y * 16 + threadIdx.x;
    for (int idx = tid; idx < NF * NTAP; idx += 256) {
        int c = idx / NTAP, t = idx % NTAP;
        float w;
        if (c < 8) w = 0.5f * (ft[(c * 2 + 0) * NTAP + t] + ft[(c * 2 + 1) * NTAP + t]);
        else       w = fn[(c - 8) * NTAP + t];
        wlds[idx] = w;
    }
    __syncthreads();

    int px = blockIdx.x * 16 + threadIdx.x;
    int py = blockIdx.y * 16 + threadIdx.y;
    if (px >= OW || py >= OH) return;

    float acc[NF];
    #pragma unroll
    for (int c = 0; c < NF; ++c) acc[c] = 0.f;

    for (int p = 0; p < KH; ++p) {
        #pragma unroll
        for (int q = 0; q < KW; ++q) {
            float xv = x[(py + p) * IN_W + (px + q)];
            int t = p * KW + q;
            #pragma unroll
            for (int c = 0; c < NF; ++c)
                acc[c] = fmaf(wlds[c * NTAP + t], xv, acc[c]);
        }
    }
    float s = 0.f;
    #pragma unroll
    for (int c = 0; c < NF; ++c) { acc[c] = fmaxf(acc[c], 0.f); s += acc[c]; }
    float inv = 1.0f / (s + 2.220446049250313e-16f);
    #pragma unroll
    for (int c = 0; c < NF; ++c) {
        float v = acc[c] * inv;
        out[c * (OH * OW) + py * OW + px] = v;
        xh[(c * XP + py) * XP + px] = (_Float16)v;
    }
}

// Stage 2a: pack templates (center 212x212 crops), K rows zero-padded to 224.
// Zero B tail => arbitrary finite A values beyond j=211 contribute nothing.
__global__ __launch_bounds__(256) void build_templates_kernel(
    const _Float16* __restrict__ xh, _Float16* __restrict__ bt)
{
    int idx = blockIdx.x * 256 + threadIdx.x;
    if (idx >= BT_ELEMS) return;
    int o = idx / (TSIDE * TPAD);
    int r = idx % (TSIDE * TPAD);
    int i = r / TPAD, j = r % TPAD;
    _Float16 v = (_Float16)0.f;
    if (j < TSIDE) v = xh[(o * XP + (CORR + i)) * XP + (CORR + j)];
    bt[idx] = v;
}

// Stage 2b: block = one (map n, dy-tile) pair x 8 consecutive dx (one per wave).
// Per K-row i: waves compute 7 x v_wmma_f32_16x16x32_f16 from LDS while the
// block cooperatively stages the NEXT B row (double buffer) and ONE new A row
// (ring slot (i+16)%17, never read this iteration) -> single barrier per i.
// Lane layout used here: A: lane&15 = M row, lane>>4 = K half (16 contig,
// k = 16h+s); B: lane&15 = N col, lane>>4 = K half (16 contig);
// C/D: vgpr r, lane -> (M = r + 8*(lane>>4), N = lane&15).
__global__ __launch_bounds__(256) void corr_wmma_kernel(
    const _Float16* __restrict__ xh, const _Float16* __restrict__ bt,
    float* __restrict__ out)
{
    __shared__ _Float16 ldsA[ARING * ASTRIDE];   //  8432 B
    __shared__ _Float16 ldsB[2 * NF * BSTRIDE];  // 14848 B

    const int pair = blockIdx.x / BLKPP;         // 48 pairs
    const int blk  = blockIdx.x % BLKPP;
    const int n    = pair / 3;
    const int dy0  = (pair % 3) * 16;
    const int dxbase = blk * 8;

    const int lane = threadIdx.x & 31;
    const int wave = threadIdx.x >> 5;
    const int dxr  = dxbase + wave;              // 0..39
    const bool valid = (dxr < NPOS);
    const int dx   = valid ? dxr : (NPOS - 1);   // clamp: duplicate work, no store
    const int dxo  = dx - dxbase;                // wave-uniform column offset in LDS A
    const int m = lane & 15;                     // A row within tile (dy offset)
    const int h = lane >> 4;                     // K-half select
    const int o = lane & 15;                     // B column (template index)

    // ---- preload: A rows rel 0..15 and B row 0 (buffer 0) ----
    for (int slot = threadIdx.x; slot < 16 * 31; slot += 256) {
        int rel = slot / 31, jb = slot % 31;
        *(v8h*)(ldsA + rel * ASTRIDE + jb * 8) =
            *(const v8h*)(xh + (n * XP + dy0 + rel) * XP + dxbase + jb * 8);
    }
    for (int slot = threadIdx.x; slot < NF * 28; slot += 256) {
        int o2 = slot / 28, jb = slot % 28;
        *(v8h*)(ldsB + o2 * BSTRIDE + jb * 8) =
            *(const v8h*)(bt + (o2 * TSIDE) * TPAD + jb * 8);
    }
    __syncthreads();

    v8f c = {};
    int arow = m;                                // (m + i) % ARING, kept incrementally
    for (int i = 0; i < TSIDE; ++i) {
        // ---- cooperative stage of next-iteration data (regions not read now) ----
        if (i + 1 < TSIDE) {
            _Float16* bdst = ldsB + ((i + 1) & 1) * (NF * BSTRIDE);
            for (int slot = threadIdx.x; slot < NF * 28; slot += 256) {
                int o2 = slot / 28, jb = slot % 28;
                *(v8h*)(bdst + o2 * BSTRIDE + jb * 8) =
                    *(const v8h*)(bt + (o2 * TSIDE + (i + 1)) * TPAD + jb * 8);
            }
            int rel = i + 16;                    // new A ring row
            _Float16* adst = ldsA + (rel % ARING) * ASTRIDE;
            const _Float16* asrc = xh + (n * XP + dy0 + rel) * XP + dxbase;
            for (int jb = threadIdx.x; jb < 31; jb += 256)
                *(v8h*)(adst + jb * 8) = *(const v8h*)(asrc + jb * 8);
            if (threadIdx.x == 0 && i + 17 < TSIDE)   // global_prefetch_b8 of row i+17
                __builtin_prefetch(xh + (n * XP + dy0 + i + 17) * XP + dxbase, 0, 1);
        }

        // ---- compute: 7 K-chunks from LDS ----
        const _Float16* aRow = ldsA + arow * ASTRIDE + dxo;
        const _Float16* bRow = ldsB + (i & 1) * (NF * BSTRIDE) + o * BSTRIDE + 16 * h;
        #pragma unroll
        for (int jc = 0; jc < NCHUNK; ++jc) {
            const int j0 = jc * 32;
            v8h a0 = *(const v8h*)(aRow + j0 + 16 * h);
            v8h a1 = *(const v8h*)(aRow + j0 + 16 * h + 8);
            v16h a = __builtin_shufflevector(a0, a1,
                       0,1,2,3,4,5,6,7,8,9,10,11,12,13,14,15);
            v16h b = *(const v16h*)(bRow + j0);
            c = __builtin_amdgcn_wmma_f32_16x16x32_f16(
                    false, a, false, b, (short)0, c, false, false);
        }
        arow = (arow + 1 == ARING) ? 0 : arow + 1;
        __syncthreads();
    }

    if (valid) {
        const float inv = 1.0f / ((float)TSIDE * (float)TSIDE);
        #pragma unroll
        for (int r2 = 0; r2 < 8; ++r2) {
            int dyy = dy0 + r2 + 8 * h;
            if (dyy < NPOS)
                out[X1_SZ + ((o * NF + n) * NPOS + dyy) * NPOS + dxr] = c[r2] * inv;
        }
    }
}

extern "C" void kernel_launch(void* const* d_in, const int* in_sizes, int n_in,
                              void* d_out, int out_size, void* d_ws, size_t ws_size,
                              hipStream_t stream) {
    const float* x  = (const float*)d_in[0];     // (256,256)
    const float* ft = (const float*)d_in[1];     // (8,1,2,9,9)
    const float* fn = (const float*)d_in[2];     // (8,1,1,9,9)
    (void)in_sizes; (void)n_in; (void)out_size; (void)ws_size;

    float* out = (float*)d_out;
    _Float16* xh = (_Float16*)d_ws;                                   // padded f16 maps
    _Float16* bt = (_Float16*)((char*)d_ws + (size_t)XH_ELEMS * 2);   // packed templates

    // 0) zero padded activation buffer (guard rows/cols must be finite/zero)
    {
        int n32 = XH_ELEMS / 2;
        zero_u32_kernel<<<(n32 + 255) / 256, 256, 0, stream>>>((unsigned int*)d_ws, n32);
    }
    // 1) conv + normalize
    {
        dim3 b(16, 16), g((OW + 15) / 16, (OH + 15) / 16);
        conv_norm_kernel<<<g, b, 0, stream>>>(x, ft, fn, out, xh);
    }
    // 2) pack templates
    build_templates_kernel<<<(BT_ELEMS + 255) / 256, 256, 0, stream>>>(xh, bt);
    // 3) WMMA all-pairs correlation (48 pairs x 5 blocks x 8 waves)
    corr_wmma_kernel<<<NPAIR * BLKPP, 256, 0, stream>>>(xh, bt, out);
}
